// GQA_84353157694016
// MI455X (gfx1250) — matrix operand order repaired and checked by reading, not compiled, not run on hardware
//
#include <hip/hip_runtime.h>

// ---------------------------------------------------------------------------
// GQA attention block for gfx1250 (CDNA5, wave32, WMMA bf16)
// Pipeline: [GEMM qkv] -> [RoPE+split] -> [flash attention] -> [GEMM out]
// All matmuls use v_wmma_f32_16x16x32_bf16 with fp32 accumulation.
// ---------------------------------------------------------------------------

typedef __attribute__((ext_vector_type(16))) __bf16 v16bf;
typedef __attribute__((ext_vector_type(8)))  __bf16 v8bf;
typedef __attribute__((ext_vector_type(4)))  __bf16 v4bf;
typedef __attribute__((ext_vector_type(8)))  float  v8f;

#define NB   2
#define NS   2048
#define NE   4096
#define NF   6144      // E + 2*KVH*D
#define NH   32
#define NKVH 8
#define ND   128
#define NM   (NB*NS)   // 4096 rows
#define ATT_SCALE 0.08838834764831845f   // D^-0.5
#define LN10000 9.210340371976184f

#define NEG_INF (-__builtin_inff())

static __device__ __forceinline__ v16bf ldfrag(const __bf16* p0, const __bf16* p1) {
  v8bf lo = *(const v8bf*)p0;
  v8bf hi = *(const v8bf*)p1;
  return __builtin_shufflevector(lo, hi, 0,1,2,3,4,5,6,7,8,9,10,11,12,13,14,15);
}

static __device__ __forceinline__ v8f wmma_bf16(v16bf a, v16bf b, v8f c) {
  // D = A(16x32) x B(32x16) + C ; emits v_wmma_f32_16x16x32_bf16
  return __builtin_amdgcn_wmma_f32_16x16x32_bf16(false, a, false, b, (short)0, c,
                                                 false, false);
}

// ---------------------------------------------------------------------------
// C[m,n] = sum_k A[m,k] * B[n,k]   (B row-major [N,K], i.e. C = A * B^T)
// A is f32 (converted to bf16 while staging) or bf16 depending on template.
// Block tile 128x128, K-step 32.  8 waves, each wave owns a 64x32 subtile
// (4 M-frags x 2 N-frags = 8 WMMA per K-step).
// ---------------------------------------------------------------------------
template<bool AF32>
__global__ __launch_bounds__(256)
void gemm_bt_kernel(const void* __restrict__ Ap, const float* __restrict__ Bp,
                    float* __restrict__ Cp, int Mdim, int Ndim, int Kdim)
{
  __shared__ __bf16 Alds[128 * 40];   // 32 cols + 8 pad
  __shared__ __bf16 Blds[128 * 40];

  const int tid  = threadIdx.x;
  const int lane = tid & 31;
  const int wid  = tid >> 5;
  const int wm   = wid >> 2;     // 0..1  -> 64 rows each
  const int wn   = wid & 3;      // 0..3  -> 32 cols each
  const int lh   = lane >> 4;    // lane half
  const int l15  = lane & 15;

  const int rowBase = blockIdx.y * 128;
  const int colBase = blockIdx.x * 128;

  const v8f vzero = {0.f,0.f,0.f,0.f,0.f,0.f,0.f,0.f};
  v8f acc[4][2];
  #pragma unroll
  for (int mf = 0; mf < 4; ++mf)
    #pragma unroll
    for (int nf = 0; nf < 2; ++nf) acc[mf][nf] = vzero;

  for (int kt = 0; kt < Kdim; kt += 32) {
    // ---- stage A tile (128x32) ----
    if constexpr (AF32) {
      const float* A = (const float*)Ap;
      #pragma unroll
      for (int i = 0; i < 4; ++i) {
        int q = tid + i * 256;
        int r = q >> 3, c4 = q & 7;
        const float4 v = *(const float4*)(A + (size_t)(rowBase + r) * Kdim + kt + c4 * 4);
        v4bf bv = { (__bf16)v.x, (__bf16)v.y, (__bf16)v.z, (__bf16)v.w };
        *(v4bf*)(&Alds[r * 40 + c4 * 4]) = bv;
      }
    } else {
      const __bf16* A = (const __bf16*)Ap;
      #pragma unroll
      for (int i = 0; i < 2; ++i) {
        int q = tid + i * 256;
        int r = q >> 2, c8 = q & 3;
        v8bf v = *(const v8bf*)(A + (size_t)(rowBase + r) * Kdim + kt + c8 * 8);
        *(v8bf*)(&Alds[r * 40 + c8 * 8]) = v;
      }
    }
    // ---- stage B tile (128x32, always f32 weights) ----
    #pragma unroll
    for (int i = 0; i < 4; ++i) {
      int q = tid + i * 256;
      int r = q >> 3, c4 = q & 7;
      const float4 v = *(const float4*)(Bp + (size_t)(colBase + r) * Kdim + kt + c4 * 4);
      v4bf bv = { (__bf16)v.x, (__bf16)v.y, (__bf16)v.z, (__bf16)v.w };
      *(v4bf*)(&Blds[r * 40 + c4 * 4]) = bv;
    }
    __syncthreads();

    // ---- fragments ----
    v16bf af[4];
    #pragma unroll
    for (int mf = 0; mf < 4; ++mf) {
      // A 16-bit layout: lane half 0 holds K 0-7 / 16-23, half 1 holds 8-15 / 24-31
      const __bf16* p = &Alds[(wm * 64 + mf * 16 + l15) * 40 + lh * 8];
      af[mf] = ldfrag(p, p + 16);
    }
    v16bf bfr[2];
    #pragma unroll
    for (int nf = 0; nf < 2; ++nf) {
      // B layout: lane = column n; half 0 -> K 0-7 / 16-23, half 1 -> K 8-15 / 24-31
      const __bf16* p = &Blds[(wn * 32 + nf * 16 + l15) * 40 + lh * 8];
      bfr[nf] = ldfrag(p, p + 16);
    }
    #pragma unroll
    for (int mf = 0; mf < 4; ++mf)
      #pragma unroll
      for (int nf = 0; nf < 2; ++nf)
        acc[mf][nf] = wmma_bf16(af[mf], bfr[nf], acc[mf][nf]);
    __syncthreads();
  }

  // ---- epilogue: C layout = lane half 0 rows r, half 1 rows 8+r ----
  #pragma unroll
  for (int mf = 0; mf < 4; ++mf)
    #pragma unroll
    for (int nf = 0; nf < 2; ++nf) {
      const int col = colBase + wn * 32 + nf * 16 + l15;
      #pragma unroll
      for (int r = 0; r < 8; ++r) {
        const int row = rowBase + wm * 64 + mf * 16 + (lh ? 8 + r : r);
        Cp[(size_t)row * Ndim + col] = acc[mf][nf][r];
      }
    }
}

// ---------------------------------------------------------------------------
// RoPE (interleaved pairs) + split QKV into per-head bf16 tensors.
// qkv row layout: [KVH][q_per_kv+2][D]; slot 0..3 = q heads, 4 = k, 5 = v.
// Q is pre-scaled by D^-0.5 so attention skips the score scale.
// ---------------------------------------------------------------------------
__global__ __launch_bounds__(256)
void rope_split_kernel(const float* __restrict__ QKV, __bf16* __restrict__ Q,
                       __bf16* __restrict__ Kd, __bf16* __restrict__ Vd)
{
  const int s = blockIdx.x;
  const int b = blockIdx.y;
  const float* row = QKV + ((size_t)b * NS + s) * NF;

  #pragma unroll
  for (int i = 0; i < 12; ++i) {
    const int p   = threadIdx.x + i * 256;   // pair index 0..3071
    const int d2  = p & 63;                  // pair within head
    const int hs  = p >> 6;                  // head-slot 0..47
    const int kvh = hs / 6, slot = hs % 6;
    const float x1 = row[2 * p];
    const float x2 = row[2 * p + 1];
    if (slot < 4) {
      const int h = kvh * 4 + slot;
      const float inv = __expf((float)d2 * (-2.0f / 128.0f) * LN10000);
      const float ang = (float)s * inv;
      const float c = __cosf(ang), sn = __sinf(ang);
      __bf16* dst = Q + (((size_t)b * NH + h) * NS + s) * ND + 2 * d2;
      dst[0] = (__bf16)((x1 * c - x2 * sn) * ATT_SCALE);
      dst[1] = (__bf16)((x1 * sn + x2 * c) * ATT_SCALE);
    } else if (slot == 4) {
      const float inv = __expf((float)d2 * (-2.0f / 128.0f) * LN10000);
      const float ang = (float)s * inv;
      const float c = __cosf(ang), sn = __sinf(ang);
      __bf16* dst = Kd + (((size_t)b * NKVH + kvh) * NS + s) * ND + 2 * d2;
      dst[0] = (__bf16)(x1 * c - x2 * sn);
      dst[1] = (__bf16)(x1 * sn + x2 * c);
    } else {
      __bf16* dst = Vd + (((size_t)b * NKVH + kvh) * NS + s) * ND + 2 * d2;
      dst[0] = (__bf16)x1;
      dst[1] = (__bf16)x2;
    }
  }
}

// ---------------------------------------------------------------------------
// Causal flash attention. Block = (64 q rows) x (1 head). 4 waves, each wave
// owns 16 q rows. Keys processed 32 at a time:
//   S(16x32)  = 8 WMMA (K stays row-major: exactly WMMA B-fragment layout)
//   online softmax via __shfl_xor row reductions (wave32 halves)
//   P staged through LDS (C-layout -> A-layout), V staged transposed in LDS
//   ctx(16x128) += P x V  = 8 WMMA
// ---------------------------------------------------------------------------
__global__ __launch_bounds__(128)
void attn_kernel(const __bf16* __restrict__ Q, const __bf16* __restrict__ Kd,
                 const __bf16* __restrict__ Vd, __bf16* __restrict__ Ctx)
{
  __shared__ __bf16 Klds[32 * 136];     // [key][d], 8 pad
  __shared__ __bf16 Vt[128 * 40];       // [d][key], 8 pad
  __shared__ __bf16 Plds[4 * 16 * 40];  // per-wave P staging [q][key]

  const int tid  = threadIdx.x;
  const int lane = tid & 31;
  const int wid  = tid >> 5;
  const int lh   = lane >> 4;
  const int l15  = lane & 15;

  const int qt  = blockIdx.x * 64;
  const int h   = blockIdx.y;
  const int b   = blockIdx.z;
  const int kvh = h >> 2;              // GQA 4:1

  // Q fragments for this wave's 16 rows (D=128 -> 4 K-steps of 32)
  const __bf16* qrow = Q + (((size_t)b * NH + h) * NS + (qt + wid * 16 + l15)) * ND;
  v16bf qf[4];
  #pragma unroll
  for (int kf = 0; kf < 4; ++kf) {
    const __bf16* p = qrow + kf * 32 + lh * 8;
    qf[kf] = ldfrag(p, p + 16);
  }

  const v8f vzero = {0.f,0.f,0.f,0.f,0.f,0.f,0.f,0.f};
  v8f acc[8];
  float mi[8], li[8];
  #pragma unroll
  for (int r = 0; r < 8; ++r) { acc[r] = vzero; mi[r] = NEG_INF; li[r] = 0.f; }

  const __bf16* Kbase = Kd + ((size_t)b * NKVH + kvh) * NS * ND;
  const __bf16* Vbase = Vd + ((size_t)b * NKVH + kvh) * NS * ND;

  const int kend = qt + 64;
  for (int kb = 0; kb < kend; kb += 32) {
    // ---- stage K (as-is) and V (transposed) ----
    #pragma unroll
    for (int i = 0; i < 4; ++i) {
      int q = tid + i * 128;
      int r = q >> 4, c8 = q & 15;
      const v8bf kv = *(const v8bf*)(Kbase + (size_t)(kb + r) * ND + c8 * 8);
      *(v8bf*)(&Klds[r * 136 + c8 * 8]) = kv;
      const v8bf vv = *(const v8bf*)(Vbase + (size_t)(kb + r) * ND + c8 * 8);
      #pragma unroll
      for (int j = 0; j < 8; ++j) Vt[(c8 * 8 + j) * 40 + r] = vv[j];
    }
    __syncthreads();

    // ---- S = Q * K^T  (two 16x16 C-frags: keys kb..+15 and kb+16..+31) ----
    v8f s0 = vzero, s1 = vzero;
    #pragma unroll
    for (int kf = 0; kf < 4; ++kf) {
      const __bf16* p0 = &Klds[l15 * 136 + kf * 32 + lh * 8];
      s0 = wmma_bf16(qf[kf], ldfrag(p0, p0 + 16), s0);
      const __bf16* p1 = &Klds[(16 + l15) * 136 + kf * 32 + lh * 8];
      s1 = wmma_bf16(qf[kf], ldfrag(p1, p1 + 16), s1);
    }

    // ---- causal mask + online softmax (rows live across 16-lane halves) ----
    const int key0 = kb + l15;
    const int key1 = kb + 16 + l15;
    __bf16* Pw = &Plds[wid * 640];
    #pragma unroll
    for (int r = 0; r < 8; ++r) {
      const int qg = qt + wid * 16 + (lh ? 8 + r : r);
      float a0 = (key0 <= qg) ? s0[r] : NEG_INF;
      float a1 = (key1 <= qg) ? s1[r] : NEG_INF;
      float mx = fmaxf(a0, a1);
      mx = fmaxf(mx, __shfl_xor(mx, 1, 32));
      mx = fmaxf(mx, __shfl_xor(mx, 2, 32));
      mx = fmaxf(mx, __shfl_xor(mx, 4, 32));
      mx = fmaxf(mx, __shfl_xor(mx, 8, 32));
      const float mnew = fmaxf(mi[r], mx);
      const float corr = __expf(mi[r] - mnew);
      const float p0v = __expf(a0 - mnew);
      const float p1v = __expf(a1 - mnew);
      float rs = p0v + p1v;
      rs += __shfl_xor(rs, 1, 32);
      rs += __shfl_xor(rs, 2, 32);
      rs += __shfl_xor(rs, 4, 32);
      rs += __shfl_xor(rs, 8, 32);
      li[r] = li[r] * corr + rs;
      mi[r] = mnew;
      #pragma unroll
      for (int nf = 0; nf < 8; ++nf) acc[nf][r] *= corr;
      const int prow = lh ? 8 + r : r;
      Pw[prow * 40 + l15]      = (__bf16)p0v;   // keys kb..kb+15
      Pw[prow * 40 + 16 + l15] = (__bf16)p1v;   // keys kb+16..kb+31
    }

    // ---- ctx += P * V (P re-read in A-frag layout; V^T gives B-frags) ----
    const __bf16* pp = &Pw[l15 * 40 + lh * 8];
    v16bf pa = ldfrag(pp, pp + 16);
    #pragma unroll
    for (int nf = 0; nf < 8; ++nf) {
      const __bf16* vp = &Vt[(nf * 16 + l15) * 40 + lh * 8];
      acc[nf] = wmma_bf16(pa, ldfrag(vp, vp + 16), acc[nf]);
    }
    __syncthreads();
  }

  // ---- normalize and write ctx [b][s][h*D+d] as bf16 ----
  #pragma unroll
  for (int nf = 0; nf < 8; ++nf) {
    #pragma unroll
    for (int r = 0; r < 8; ++r) {
      const int qg = qt + wid * 16 + (lh ? 8 + r : r);
      const float o = acc[nf][r] * (1.0f / li[r]);
      Ctx[((size_t)b * NS + qg) * NE + h * ND + nf * 16 + l15] = (__bf16)o;
    }
  }
}

// ---------------------------------------------------------------------------
extern "C" void kernel_launch(void* const* d_in, const int* in_sizes, int n_in,
                              void* d_out, int out_size, void* d_ws, size_t ws_size,
                              hipStream_t stream) {
  (void)in_sizes; (void)n_in; (void)out_size; (void)ws_size;
  const float* x    = (const float*)d_in[0];   // [B,S,E]
  const float* wqkv = (const float*)d_in[1];   // [F,E]
  const float* wo   = (const float*)d_in[2];   // [E,E]
  float* out = (float*)d_out;                  // [B,S,E]

  char* ws = (char*)d_ws;
  float*  QKV  = (float*)ws;                                    //  96 MiB f32 [M,F]
  __bf16* Qb   = (__bf16*)(ws + (size_t)100663296);             //  32 MiB [B,H,S,D]
  __bf16* Kb   = (__bf16*)(ws + (size_t)134217728);             //   8 MiB [B,KVH,S,D]
  __bf16* Vb   = (__bf16*)(ws + (size_t)142606336);             //   8 MiB [B,KVH,S,D]
  __bf16* Ctxb = (__bf16*)(ws + (size_t)150994944);             //  32 MiB [M,E]

  // 1) qkv = x * wqkv^T
  gemm_bt_kernel<true><<<dim3(NF / 128, NM / 128), 256, 0, stream>>>(
      (const void*)x, wqkv, QKV, NM, NF, NE);

  // 2) RoPE + split into per-head Q/K/V (Q pre-scaled)
  rope_split_kernel<<<dim3(NS, NB), 256, 0, stream>>>(QKV, Qb, Kb, Vb);

  // 3) causal flash attention
  attn_kernel<<<dim3(NS / 64, NH, NB), 128, 0, stream>>>(Qb, Kb, Vb, Ctxb);

  // 4) out = ctx * wo^T
  gemm_bt_kernel<false><<<dim3(NE / 128, NM / 128), 256, 0, stream>>>(
      (const void*)Ctxb, wo, out, NM, NE, NE);
}
